// MLA_292057776688
// MI455X (gfx1250) — matrix operand order repaired and checked by reading, not compiled, not run on hardware
//
#include <hip/hip_runtime.h>
#include <hip/hip_bf16.h>
#include <math.h>

// ---------------------------------------------------------------------------
// Problem constants (match reference setup_inputs)
// ---------------------------------------------------------------------------
#define BB 2
#define SS 2048
#define DD 1024
#define HH 16
#define PH 64
#define MM (BB * SS)          // 4096 rows
#define NQKV (DD + DD + PH)   // 2112
#define NKV  (2 * DD)         // 2048

typedef __bf16 bf16;
typedef __attribute__((ext_vector_type(16))) __bf16 v16bf;
typedef __attribute__((ext_vector_type(8)))  float  v8f;

#define NEG_MAX (-3.402823466e38f)
#define RMS_EPS (1.1920928955078125e-07f)

// ---------------------------------------------------------------------------
// CDNA5 async global->LDS copy (ASYNCcnt-tracked). LDS destination address is
// per-lane from a VGPR; flat pointers to __shared__ carry the DS-space offset
// in bits [31:0] (ISA 10.2 aperture mapping), so truncation gives the ds addr.
// ---------------------------------------------------------------------------
__device__ __forceinline__ unsigned lds_u32(const void* p) {
    return (unsigned)(unsigned long long)(size_t)p;
}
__device__ __forceinline__ void async_copy_b128(unsigned lds_byte, const void* gp) {
    asm volatile("global_load_async_to_lds_b128 %0, %1, off"
                 :: "v"(lds_byte), "v"((unsigned long long)(size_t)gp)
                 : "memory");
}
__device__ __forceinline__ void wait_async0() {
    asm volatile("s_wait_asynccnt 0x0" ::: "memory");
}
// previous tile (3 older of <=6 outstanding ops) complete; ASYNCcnt is in-order
__device__ __forceinline__ void wait_async3() {
    asm volatile("s_wait_asynccnt 0x3" ::: "memory");
}

// ===========================================================================
// Generic GEMM:  C[M,N] = A[M,K] @ W[N,K]^T + bias[N]
// bf16 in / fp32 out, bf16 WMMA compute, double-buffered async global->LDS
// staging (DMA of tile i+1 overlaps WMMA on tile i).
// Block: 256 threads (8 waves). Block tile 128x64, wave tile 32x32, K-step 32.
// ===========================================================================
__global__ __launch_bounds__(256) void gemm_bf16(
    const __bf16* __restrict__ A, const __bf16* __restrict__ W,
    const float* __restrict__ bias, float* __restrict__ C,
    int M, int N, int K)
{
    __shared__ __bf16 As[2][128][32];   // 2 x 8 KB
    __shared__ __bf16 Bs[2][64][32];    // 2 x 4 KB

    const int t    = threadIdx.x;
    const int lane = t & 31;
    const int w    = t >> 5;
    const int wm   = w & 3;      // row group (4 x 32 = 128)
    const int wn   = w >> 2;     // col group (2 x 32 = 64)
    const int lr   = lane & 15;
    const int hs   = lane >> 4;  // half-wave select
    const int m0   = blockIdx.x * 128;
    const int n0   = blockIdx.y * 64;

    // staging assignment: 16B chunks; A tile = 512 chunks (2/thread), B = 256.
    const int arow = t >> 2;          // 0..63
    const int aseg = t & 3;           // 0..3 (8 bf16 elems each)
    unsigned aoff[2], boff[2];
    #pragma unroll
    for (int bufi = 0; bufi < 2; ++bufi) {
        aoff[bufi] = lds_u32(&As[bufi][0][0]) + (unsigned)(arow * 64 + aseg * 16);
        boff[bufi] = lds_u32(&Bs[bufi][0][0]) + (unsigned)(arow * 64 + aseg * 16);
    }
    const __bf16* agp0 = A + (size_t)(m0 + arow) * K + aseg * 8;
    const __bf16* agp1 = A + (size_t)(m0 + 64 + arow) * K + aseg * 8;
    const __bf16* wgp  = W + (size_t)(n0 + arow) * K + aseg * 8;

    v8f acc[2][2] = {};

    const int nIter = K >> 5;
    // prologue: stage tile 0 into buffer 0
    async_copy_b128(aoff[0],           agp0);
    async_copy_b128(aoff[0] + 64 * 64, agp1);
    async_copy_b128(boff[0],           wgp);

    for (int it = 0; it < nIter; ++it) {
        const int cur = it & 1;
        if (it + 1 < nIter) {
            const int nk = (it + 1) << 5;
            async_copy_b128(aoff[cur ^ 1],           agp0 + nk);
            async_copy_b128(aoff[cur ^ 1] + 64 * 64, agp1 + nk);
            async_copy_b128(boff[cur ^ 1],           wgp  + nk);
            wait_async3();                 // tile `it` landed; tile it+1 in flight
        } else {
            wait_async0();
        }
        __syncthreads();

        // ---- fragments + WMMA on buffer `cur` ----
        v16bf afrag[2], bfrag[2];
        #pragma unroll
        for (int mt = 0; mt < 2; ++mt) {
            const __bf16* src = &As[cur][wm * 32 + mt * 16 + lr][hs * 8];
            #pragma unroll
            for (int i = 0; i < 8; ++i) { afrag[mt][i] = src[i]; afrag[mt][8 + i] = src[16 + i]; }
        }
        #pragma unroll
        for (int nt = 0; nt < 2; ++nt) {
            const __bf16* src = &Bs[cur][wn * 32 + nt * 16 + lr][hs * 8];
            #pragma unroll
            for (int i = 0; i < 8; ++i) { bfrag[nt][i] = src[i]; bfrag[nt][8 + i] = src[16 + i]; }
        }
        #pragma unroll
        for (int mt = 0; mt < 2; ++mt)
            #pragma unroll
            for (int nt = 0; nt < 2; ++nt)
                acc[mt][nt] = __builtin_amdgcn_wmma_f32_16x16x32_bf16(
                    false, afrag[mt], false, bfrag[nt], (short)0, acc[mt][nt], false, false);
        __syncthreads();                   // reads done before buffer reuse
    }

    // ---- epilogue: bias + store (C layout: M = vgpr + 8*hs, N = lr) ----
    #pragma unroll
    for (int mt = 0; mt < 2; ++mt)
        #pragma unroll
        for (int nt = 0; nt < 2; ++nt) {
            int n = n0 + wn * 32 + nt * 16 + lr;
            float bv = bias[n];
            #pragma unroll
            for (int vi = 0; vi < 8; ++vi) {
                int m = m0 + wm * 32 + mt * 16 + vi + hs * 8;
                C[(size_t)m * N + n] = acc[mt][nt][vi] + bv;
            }
        }
}

// ===========================================================================
// fp32 -> bf16 conversion (weights / embedding), n multiple of 4
// ===========================================================================
__global__ __launch_bounds__(256) void cvt_bf16_kernel(
    const float* __restrict__ in, __bf16* __restrict__ out, size_t n)
{
    size_t i = ((size_t)blockIdx.x * 256 + threadIdx.x) * 4;
    if (i < n) {
        float4 f = *(const float4*)(in + i);
        out[i + 0] = (__bf16)f.x;
        out[i + 1] = (__bf16)f.y;
        out[i + 2] = (__bf16)f.z;
        out[i + 3] = (__bf16)f.w;
    }
}

// ===========================================================================
// RMS-norm of kv_lora = qkv[:, 1024:2048] -> kvln (bf16, feeds kv GEMM)
// ===========================================================================
__global__ __launch_bounds__(256) void kvln_kernel(
    const float* __restrict__ qkv, const float* __restrict__ g,
    __bf16* __restrict__ out)
{
    const int row = blockIdx.x;
    const float* x = qkv + (size_t)row * NQKV + DD;
    float xl[4]; float s = 0.f;
    #pragma unroll
    for (int i = 0; i < 4; ++i) { xl[i] = x[threadIdx.x + i * 256]; s += xl[i] * xl[i]; }
    #pragma unroll
    for (int off = 16; off; off >>= 1) s += __shfl_xor(s, off, 32);
    __shared__ float red[8];
    if ((threadIdx.x & 31) == 0) red[threadIdx.x >> 5] = s;
    __syncthreads();
    float tot = 0.f;
    #pragma unroll
    for (int i = 0; i < 8; ++i) tot += red[i];
    float rs = rsqrtf(tot * (1.f / (float)DD) + RMS_EPS);
    __bf16* o = out + (size_t)row * DD;
    #pragma unroll
    for (int i = 0; i < 4; ++i) { int c = threadIdx.x + i * 256; o[c] = (__bf16)(xl[i] * rs * g[c]); }
}

// ===========================================================================
// Per-(b,s,h): q = rmsnorm(q_lin @ Wq^T + bq, g_q); rope(rot=32) on [64:96);
// scale D^-0.5 ; write bf16 qh[(b*H+h)*S+s][128].  64 threads, grid = M*H
// ===========================================================================
__global__ __launch_bounds__(64) void qproj_kernel(
    const float* __restrict__ qkv, const float* __restrict__ Wq,
    const float* __restrict__ bq, const float* __restrict__ gq,
    __bf16* __restrict__ qh)
{
    const int h  = blockIdx.x & (HH - 1);
    const int bs = blockIdx.x >> 4;            // = b*S + s = qkv row
    const int s  = bs & (SS - 1);
    const int b  = bs >> 11;
    const int t  = threadIdx.x;

    __shared__ float xv[PH];
    __shared__ float qn[2 * PH];
    __shared__ float red2[2];

    xv[t] = qkv[(size_t)bs * NQKV + h * PH + t];
    __syncthreads();

    float o0 = bq[t], o1 = bq[t + 64];
    #pragma unroll 8
    for (int k = 0; k < PH; ++k) {
        float x = xv[k];
        o0 += x * Wq[t * PH + k];
        o1 += x * Wq[(t + 64) * PH + k];
    }
    float ss = o0 * o0 + o1 * o1;
    #pragma unroll
    for (int off = 16; off; off >>= 1) ss += __shfl_xor(ss, off, 32);
    if ((t & 31) == 0) red2[t >> 5] = ss;
    __syncthreads();
    float rs = rsqrtf((red2[0] + red2[1]) * (1.f / 128.f) + RMS_EPS);
    qn[t]      = o0 * rs * gq[t];
    qn[t + 64] = o1 * rs * gq[t + 64];
    __syncthreads();

    const float scale = 0.03125f;  // 1024^-0.5
    size_t base = (((size_t)b * HH + h) * SS + s) * 128;
    #pragma unroll
    for (int half = 0; half < 2; ++half) {
        int o = t + half * 64;
        float v = qn[o];
        if (o >= 64 && o < 96) {       // rope region: first 32 of rope half
            int r = o - 64;
            int p = r >> 1;
            float inv = __expf(-9.210340371976184f * (float)p * (1.f / 16.f)); // 1e4^(-2p/32)
            float fr = (float)s * inv;
            float c = __cosf(fr), sn = __sinf(fr);
            float partner = qn[o ^ 1];
            v = (r & 1) ? (v * c + partner * sn) : (v * c - partner * sn);
        }
        qh[base + o] = (__bf16)(v * scale);
    }
}

// ===========================================================================
// Pack kh = [k_nope | k_rope] (bf16, B,H,S,128) and vT (bf16, B,H,64,S)
// ===========================================================================
__global__ __launch_bounds__(256) void pack_kernel(
    const float* __restrict__ qkv, const float* __restrict__ kv,
    __bf16* __restrict__ kh, __bf16* __restrict__ vT,
    size_t totKh, size_t totVT)
{
    size_t i = (size_t)blockIdx.x * 256 + threadIdx.x;
    if (i < totKh) {
        int d = (int)(i & 127);
        size_t r = i >> 7;                 // (b*H+h)*S + s
        int s = (int)(r & (SS - 1));
        size_t bh = r >> 11;
        int h = (int)(bh & (HH - 1));
        int b = (int)(bh >> 4);
        size_t row = (size_t)b * SS + s;
        float val = (d < 64) ? kv[row * NKV + h * 128 + d]
                             : qkv[row * NQKV + 2 * DD + (d - 64)];
        kh[i] = (__bf16)val;
    } else if (i < totKh + totVT) {
        size_t j = i - totKh;              // layout (b,h,d,s), s fastest
        int s = (int)(j & (SS - 1));
        size_t r = j >> 11;
        int d = (int)(r & 63);
        size_t bh = r >> 6;
        int h = (int)(bh & (HH - 1));
        int b = (int)(bh >> 4);
        size_t row = (size_t)b * SS + s;
        vT[j] = (__bf16)kv[row * NKV + h * 128 + 64 + d];
    }
}

// ===========================================================================
// Flash attention. 128 threads (4 waves); wave = one 16-row q tile;
// block = 64 q rows of one (b,h). Key tiles of 32 (causal-trimmed).
// sim: 8 x wmma_bf16 (K=128); P@V: 4 x wmma_bf16 (P staged through LDS).
// V is pre-transposed (B,H,64,S) so B-operand loads are two contiguous 16B runs.
// Output: attn (B,S,H*64) bf16 (feeds final GEMM).
// ===========================================================================
__global__ __launch_bounds__(128) void attn_kernel(
    const __bf16* __restrict__ qh, const __bf16* __restrict__ kh,
    const __bf16* __restrict__ vT, const int* __restrict__ past_p,
    __bf16* __restrict__ attn)
{
    const int past = past_p[0];
    const int wave = threadIdx.x >> 5;
    const int lane = threadIdx.x & 31;
    const int lr   = lane & 15;
    const int hs   = lane >> 4;
    const int qt   = blockIdx.x & (SS / 64 - 1);
    const int bh   = blockIdx.x >> 5;           // / (S/64) = /32
    const int qs0  = qt * 64 + wave * 16;

    const __bf16* qrow  = qh + (size_t)bh * SS * 128;
    const __bf16* krow  = kh + (size_t)bh * SS * 128;
    const __bf16* vtrow = vT + (size_t)bh * 64 * SS;

    // load Q fragments: 4 K-chunks of 32
    v16bf qa[4];
    {
        const __bf16* src = qrow + (size_t)(qs0 + lr) * 128 + hs * 8;
        #pragma unroll
        for (int kc = 0; kc < 4; ++kc)
            #pragma unroll
            for (int i = 0; i < 8; ++i) {
                qa[kc][i]     = src[kc * 32 + i];
                qa[kc][8 + i] = src[kc * 32 + 16 + i];
            }
    }

    float mrun[8], lrun[8];
    v8f o[4] = {};
    #pragma unroll
    for (int i = 0; i < 8; ++i) { mrun[i] = NEG_MAX; lrun[i] = 0.f; }

    __shared__ __bf16 Pb[4][16][32];

    int jneed = qs0 + 15 + past;                  // keys j < i+past valid
    int ntiles = (jneed + 31) >> 5;
    if (ntiles < 1) ntiles = 1;
    if (ntiles > SS / 32) ntiles = SS / 32;

    for (int jt = 0; jt < ntiles; ++jt) {
        const int j0 = jt * 32;

        // ---- sim tile (16 q x 32 k) ----
        v8f sfrag[2] = {};
        #pragma unroll
        for (int half = 0; half < 2; ++half) {
            const __bf16* kb = krow + (size_t)(j0 + half * 16 + lr) * 128 + hs * 8;
            #pragma unroll
            for (int kc = 0; kc < 4; ++kc) {
                v16bf bf;
                #pragma unroll
                for (int i = 0; i < 8; ++i) { bf[i] = kb[kc * 32 + i]; bf[8 + i] = kb[kc * 32 + 16 + i]; }
                sfrag[half] = __builtin_amdgcn_wmma_f32_16x16x32_bf16(
                    false, qa[kc], false, bf, (short)0, sfrag[half], false, false);
            }
        }

        // ---- mask + online softmax (per-row reductions over 16 lanes) ----
        float p0[8], p1[8];
        #pragma unroll
        for (int vi = 0; vi < 8; ++vi) {
            int m = vi + hs * 8;
            int qi = qs0 + m;
            float s0 = ((j0 + lr)      >= qi + past) ? NEG_MAX : sfrag[0][vi];
            float s1 = ((j0 + 16 + lr) >= qi + past) ? NEG_MAX : sfrag[1][vi];
            float mx = fmaxf(s0, s1);
            #pragma unroll
            for (int off = 1; off < 16; off <<= 1) mx = fmaxf(mx, __shfl_xor(mx, off, 32));
            float newm = fmaxf(mrun[vi], mx);
            float a  = __expf(mrun[vi] - newm);
            float e0 = __expf(s0 - newm);
            float e1 = __expf(s1 - newm);
            float rsum = e0 + e1;
            #pragma unroll
            for (int off = 1; off < 16; off <<= 1) rsum += __shfl_xor(rsum, off, 32);
            lrun[vi] = lrun[vi] * a + rsum;
            mrun[vi] = newm;
            p0[vi] = e0; p1[vi] = e1;
            #pragma unroll
            for (int nt = 0; nt < 4; ++nt) o[nt][vi] *= a;
        }

        // ---- stage P (16x32 bf16) through per-wave LDS, reload as A-frag ----
        #pragma unroll
        for (int vi = 0; vi < 8; ++vi) {
            int m = vi + hs * 8;
            Pb[wave][m][lr]      = (__bf16)p0[vi];
            Pb[wave][m][16 + lr] = (__bf16)p1[vi];
        }
        v16bf pa;
        {
            const __bf16* src = &Pb[wave][lr][hs * 8];
            #pragma unroll
            for (int i = 0; i < 8; ++i) { pa[i] = src[i]; pa[8 + i] = src[16 + i]; }
        }

        // ---- O += P @ V (B-operand: two 16B runs of vT row, k = 8h+s / 16+8h+s) ----
        #pragma unroll
        for (int nt = 0; nt < 4; ++nt) {
            v16bf vb;
            const __bf16* src = vtrow + (size_t)(nt * 16 + lr) * SS + j0 + hs * 8;
            #pragma unroll
            for (int i = 0; i < 8; ++i) { vb[i] = src[i]; vb[8 + i] = src[16 + i]; }
            o[nt] = __builtin_amdgcn_wmma_f32_16x16x32_bf16(
                false, pa, false, vb, (short)0, o[nt], false, false);
        }
    }

    // ---- epilogue: normalize, write (B,S,H*64) bf16 ----
    const int b = bh >> 4, h = bh & (HH - 1);
    #pragma unroll
    for (int vi = 0; vi < 8; ++vi) {
        int m = vi + hs * 8;
        float inv_l = 1.f / fmaxf(lrun[vi], 1e-30f);
        size_t base = ((size_t)b * SS + qs0 + m) * (size_t)DD + h * 64;
        #pragma unroll
        for (int nt = 0; nt < 4; ++nt)
            attn[base + nt * 16 + lr] = (__bf16)(o[nt][vi] * inv_l);
    }
}

// ===========================================================================
// Launch
// ===========================================================================
extern "C" void kernel_launch(void* const* d_in, const int* in_sizes, int n_in,
                              void* d_out, int out_size, void* d_ws, size_t ws_size,
                              hipStream_t stream) {
    const float* emb  = (const float*)d_in[0];
    const int*   past = (const int*)  d_in[1];
    const float* Wqkv = (const float*)d_in[2];
    const float* bqkv = (const float*)d_in[3];
    const float* Wq   = (const float*)d_in[4];
    const float* bq   = (const float*)d_in[5];
    const float* gq   = (const float*)d_in[6];
    const float* gkv  = (const float*)d_in[7];
    const float* Wkv  = (const float*)d_in[8];
    const float* bkv  = (const float*)d_in[9];
    const float* Wout = (const float*)d_in[10];
    const float* bout = (const float*)d_in[11];
    float* out = (float*)d_out;

    char* ws = (char*)d_ws;
    size_t off = 0;
    float*  qkv    = (float*)(ws + off);  off += (size_t)MM * NQKV * 4;
    float*  kv     = (float*)(ws + off);  off += (size_t)MM * NKV * 4;
    __bf16* embbf  = (__bf16*)(ws + off); off += (size_t)MM * DD * 2;
    __bf16* wqkvbf = (__bf16*)(ws + off); off += (size_t)NQKV * DD * 2;
    __bf16* wkvbf  = (__bf16*)(ws + off); off += (size_t)NKV * DD * 2;
    __bf16* woutbf = (__bf16*)(ws + off); off += (size_t)DD * DD * 2;
    __bf16* kvlnbf = (__bf16*)(ws + off); off += (size_t)MM * DD * 2;
    __bf16* qh     = (__bf16*)(ws + off); off += (size_t)BB * HH * SS * 128 * 2;
    __bf16* kh     = (__bf16*)(ws + off); off += (size_t)BB * HH * SS * 128 * 2;
    __bf16* vT     = (__bf16*)(ws + off); off += (size_t)BB * HH * 64 * SS * 2;
    __bf16* attnbf = (__bf16*)(ws + off); off += (size_t)MM * DD * 2;

    // 0) pre-convert A / W operands to bf16
    {
        size_t n;
        n = (size_t)MM * DD;     cvt_bf16_kernel<<<(unsigned)(n / 1024), 256, 0, stream>>>(emb,  embbf,  n);
        n = (size_t)NQKV * DD;   cvt_bf16_kernel<<<(unsigned)(n / 1024), 256, 0, stream>>>(Wqkv, wqkvbf, n);
        n = (size_t)NKV * DD;    cvt_bf16_kernel<<<(unsigned)(n / 1024), 256, 0, stream>>>(Wkv,  wkvbf,  n);
        n = (size_t)DD * DD;     cvt_bf16_kernel<<<(unsigned)(n / 1024), 256, 0, stream>>>(Wout, woutbf, n);
    }
    // 1) qkv = emb @ Wqkv^T + bqkv
    gemm_bf16<<<dim3(MM / 128, NQKV / 64), 256, 0, stream>>>(embbf, wqkvbf, bqkv, qkv, MM, NQKV, DD);
    // 2) kvln = rmsnorm(qkv[:,1024:2048]) * g_kv   (bf16)
    kvln_kernel<<<MM, 256, 0, stream>>>(qkv, gkv, kvlnbf);
    // 3) kv = kvln @ Wkv^T + bkv
    gemm_bf16<<<dim3(MM / 128, NKV / 64), 256, 0, stream>>>(kvlnbf, wkvbf, bkv, kv, MM, NKV, DD);
    // 4) qh (bf16, scaled, rope'd)
    qproj_kernel<<<MM * HH, 64, 0, stream>>>(qkv, Wq, bq, gq, qh);
    // 5) pack kh / vT
    size_t totKh = (size_t)BB * HH * SS * 128, totVT = (size_t)BB * HH * 64 * SS;
    pack_kernel<<<(unsigned)((totKh + totVT + 255) / 256), 256, 0, stream>>>(qkv, kv, kh, vT, totKh, totVT);
    // 6) attention -> attn bf16
    attn_kernel<<<BB * HH * (SS / 64), 128, 0, stream>>>(qh, kh, vT, past, attnbf);
    // 7) out = attn @ Wout^T + bout
    gemm_bf16<<<dim3(MM / 128, DD / 64), 256, 0, stream>>>(attnbf, woutbf, bout, out, MM, DD, DD);
}
